// Block_68719476955
// MI455X (gfx1250) — matrix operand order repaired and checked by reading, not compiled, not run on hardware
//
#include <hip/hip_runtime.h>
#include <hip/hip_bf16.h>

typedef __bf16 bf16;
typedef __attribute__((ext_vector_type(16))) __bf16 v16bf;
typedef __attribute__((ext_vector_type(8)))  __bf16 v8bf;
typedef __attribute__((ext_vector_type(4)))  __bf16 v4bf;
typedef __attribute__((ext_vector_type(8)))  float  v8f;
typedef __attribute__((ext_vector_type(4)))  float  v4f;

#define B_DIM   4
#define S_DIM   2048
#define E_DIM   2048
#define M_ROWS  (B_DIM * S_DIM)   // 8192 rows
#define FF_DIM  (4 * E_DIM)       // 8192
#define LN_EPS  1e-5f

// ---------------------------------------------------------------------------
// CDNA5 async global->LDS copy (ASYNCcnt path) + wait, via inline asm so it
// works on both toolchains regardless of builtin arity.
// ---------------------------------------------------------------------------
__device__ __forceinline__ void async_ld16(unsigned lds_addr, const void* gptr) {
    asm volatile("global_load_async_to_lds_b128 %0, %1, off"
                 :: "v"(lds_addr), "v"((unsigned long long)(size_t)gptr)
                 : "memory");
}
__device__ __forceinline__ void wait_async0() {
    asm volatile("s_wait_asynccnt 0" ::: "memory");
}

// ---------------------------------------------------------------------------
// f32 -> bf16 elementwise converter (vectorized: float4 load, 4x bf16 store)
// ---------------------------------------------------------------------------
__global__ void cvt_f32_bf16(const float* __restrict__ in, bf16* __restrict__ out, int n4) {
    int i = blockIdx.x * blockDim.x + threadIdx.x;
    if (i < n4) {
        v4f v = ((const v4f*)in)[i];
        ((v4bf*)out)[i] = __builtin_convertvector(v, v4bf);
    }
}

// ---------------------------------------------------------------------------
// Transpose + convert: w (K x N, f32, row-major) -> wT (N x K, bf16, row-major)
// ---------------------------------------------------------------------------
__global__ void transpose_cvt(const float* __restrict__ w, bf16* __restrict__ wT,
                              int K, int N) {
    __shared__ float tile[32][33];
    const int n0 = blockIdx.x * 32;
    const int k0 = blockIdx.y * 32;
    const int c  = threadIdx.x & 31;
    const int r0 = threadIdx.x >> 5;   // 0..7
#pragma unroll
    for (int rr = 0; rr < 4; ++rr) {
        int r = r0 * 4 + rr;
        tile[r][c] = w[(size_t)(k0 + r) * N + n0 + c];
    }
    __syncthreads();
#pragma unroll
    for (int rr = 0; rr < 4; ++rr) {
        int n = r0 * 4 + rr;
        wT[(size_t)(n0 + n) * K + k0 + c] = (bf16)tile[c][n];
    }
}

// ---------------------------------------------------------------------------
// bf16 WMMA GEMM: C = A(MxK) * BT(NxK)^T, f32 accumulate.
// EPI==0: outF = acc + bias[n] + resid      (f32)
// EPI==1: outB = gelu_exact(acc + bias[n])  (bf16)
//
// Block = 256 threads (8 waves). Block tile 256x128, BK=32.
// Waves in 4(M) x 2(N) grid; each wave owns 64x64 => 4x4 accumulators of
// 16x16 => 16 v_wmma per K-step vs 16 ds_load_b128 (1:1 ratio).
// Tiles staged into double-buffered LDS with global_load_async_to_lds_b128,
// overlapped with WMMA compute; one s_wait_asynccnt + barrier per stage.
// ---------------------------------------------------------------------------
#define BM 256
#define BN 128
#define BK 32
#define LDS_ROW 80                     // 32 bf16 = 64B + 16B pad (conflict-free)
#define SZ_A   (BM * LDS_ROW)          // 20480
#define SZ_B   (BN * LDS_ROW)          // 10240
#define SZ_STG (SZ_A + SZ_B)           // 30720; x2 stages = 61440 B

template <int EPI>
__launch_bounds__(256)
__global__ void gemm_bf16(const bf16* __restrict__ A, const bf16* __restrict__ BT,
                          const float* __restrict__ bias, const float* __restrict__ resid,
                          float* __restrict__ outF, bf16* __restrict__ outB,
                          int M, int N, int K) {
    __shared__ alignas(16) unsigned char smem[2 * SZ_STG];

    const int tid  = threadIdx.x;
    const int lane = tid & 31;
    const int wave = tid >> 5;
    const int wm   = wave >> 1;   // 0..3
    const int wn   = wave & 1;    // 0..1
    const int tileM = blockIdx.y * BM;
    const int tileN = blockIdx.x * BN;

    // ---- async tile-load mapping --------------------------------------
    // A stage: 256 rows x 32 cols bf16; one thread per row, 64B = 4x b128.
    // B stage: 128 rows x 32 cols bf16; two threads per row, 32B = 2x b128.
    const unsigned smbase = (unsigned)(size_t)smem;      // LDS byte offset
    const unsigned aoff   = smbase + tid * LDS_ROW;
    const unsigned boff   = smbase + SZ_A + (tid >> 1) * LDS_ROW + (tid & 1) * 32;
    const bf16* gA = A  + (size_t)(tileM + tid) * K;
    const bf16* gB = BT + (size_t)(tileN + (tid >> 1)) * K + (tid & 1) * 16;

    auto issue = [&](int kt, int s) {
        const bf16* pa = gA + kt * BK;
        const bf16* pb = gB + kt * BK;
        unsigned la = aoff + s * SZ_STG;
        unsigned lb = boff + s * SZ_STG;
        async_ld16(la,      pa);
        async_ld16(la + 16, pa + 8);
        async_ld16(la + 32, pa + 16);
        async_ld16(la + 48, pa + 24);
        async_ld16(lb,      pb);
        async_ld16(lb + 16, pb + 8);
    };

    v8f acc[4][4] = {};

    auto compute = [&](int s) {
        const unsigned char* sA = smem + s * SZ_STG;
        const unsigned char* sB = sA + SZ_A;
        union { v16bf v; v8bf h[2]; } af[4], bfg[4];
        // A fragment: lane row m = lane&15; K base 0/8 elems (lo/hi halfwave):
        // two 16B runs at +0/+32 bytes within the 32-K slice.
#pragma unroll
        for (int mi = 0; mi < 4; ++mi) {
            int off = (wm * 64 + mi * 16 + (lane & 15)) * LDS_ROW + ((lane >> 4) << 4);
            af[mi].h[0] = *(const v8bf*)(sA + off);
            af[mi].h[1] = *(const v8bf*)(sA + off + 32);
        }
        // B fragment: lane col n = lane&15; K base 0/8 elems (lo/hi halfwave):
        // two 16B runs at +0/+32 bytes within the 32-K slice.
#pragma unroll
        for (int ni = 0; ni < 4; ++ni) {
            int off = (wn * 64 + ni * 16 + (lane & 15)) * LDS_ROW + ((lane >> 4) << 4);
            bfg[ni].h[0] = *(const v8bf*)(sB + off);
            bfg[ni].h[1] = *(const v8bf*)(sB + off + 32);
        }
#pragma unroll
        for (int mi = 0; mi < 4; ++mi)
#pragma unroll
            for (int ni = 0; ni < 4; ++ni)
                acc[mi][ni] = __builtin_amdgcn_wmma_f32_16x16x32_bf16(
                    false, af[mi].v, false, bfg[ni].v,
                    (short)0, acc[mi][ni], false, false);
    };

    const int nk = K >> 5;     // K / 32
    issue(0, 0);
    wait_async0();
    __syncthreads();
    for (int kt = 0; kt < nk; ++kt) {
        const int  s    = kt & 1;
        const bool more = (kt + 1) < nk;
        if (more) issue(kt + 1, s ^ 1);  // DMA next stage during WMMAs
        compute(s);
        if (more) {
            wait_async0();
            __syncthreads();
        }
    }

    // Epilogue. C layout: VGPR v, lane l -> M = v + 8*(l>=16), N = l&15.
#pragma unroll
    for (int mi = 0; mi < 4; ++mi) {
        const int rbase = tileM + wm * 64 + mi * 16 + ((lane >> 4) << 3);
#pragma unroll
        for (int ni = 0; ni < 4; ++ni) {
            const int c  = tileN + wn * 64 + ni * 16 + (lane & 15);
            const float bb = bias[c];
#pragma unroll
            for (int v = 0; v < 8; ++v) {
                const size_t idx = (size_t)(rbase + v) * N + c;
                float t = acc[mi][ni][v] + bb;
                if constexpr (EPI == 1) {
                    float gel = 0.5f * t * (1.0f + erff(t * 0.70710678118654752f));
                    outB[idx] = (bf16)gel;
                } else {
                    outF[idx] = t + resid[idx];
                }
            }
        }
    }
}

// ---------------------------------------------------------------------------
// Fused LayerNorm over E=2048 per row. One 256-thread block per row.
// ---------------------------------------------------------------------------
template <bool BF16OUT>
__global__ void ln_kernel(const float* __restrict__ in, const float* __restrict__ g,
                          const float* __restrict__ bta, float* __restrict__ outF,
                          bf16* __restrict__ outB) {
    __shared__ float redS[8];
    __shared__ float redQ[8];
    const int row = blockIdx.x;
    const int tid = threadIdx.x;
    const size_t base = (size_t)row * E_DIM + tid * 8;

    v4f a0 = ((const v4f*)(in + base))[0];
    v4f a1 = ((const v4f*)(in + base))[1];
    float s = a0.x + a0.y + a0.z + a0.w + a1.x + a1.y + a1.z + a1.w;
    float q = a0.x*a0.x + a0.y*a0.y + a0.z*a0.z + a0.w*a0.w
            + a1.x*a1.x + a1.y*a1.y + a1.z*a1.z + a1.w*a1.w;

#pragma unroll
    for (int off = 16; off > 0; off >>= 1) {
        s += __shfl_xor(s, off, 32);
        q += __shfl_xor(q, off, 32);
    }
    if ((tid & 31) == 0) { redS[tid >> 5] = s; redQ[tid >> 5] = q; }
    __syncthreads();
    s = 0.f; q = 0.f;
#pragma unroll
    for (int i = 0; i < 8; ++i) { s += redS[i]; q += redQ[i]; }

    const float mean = s * (1.0f / E_DIM);
    const float var  = q * (1.0f / E_DIM) - mean * mean;
    const float rstd = rsqrtf(var + LN_EPS);

    v4f g0 = ((const v4f*)(g   + tid * 8))[0];
    v4f g1 = ((const v4f*)(g   + tid * 8))[1];
    v4f b0 = ((const v4f*)(bta + tid * 8))[0];
    v4f b1 = ((const v4f*)(bta + tid * 8))[1];
    v4f o0 = (a0 - mean) * rstd * g0 + b0;
    v4f o1 = (a1 - mean) * rstd * g1 + b1;

    if constexpr (BF16OUT) {
        ((v4bf*)(outB + base))[0] = __builtin_convertvector(o0, v4bf);
        ((v4bf*)(outB + base))[1] = __builtin_convertvector(o1, v4bf);
    } else {
        ((v4f*)(outF + base))[0] = o0;
        ((v4f*)(outF + base))[1] = o1;
    }
}

// ---------------------------------------------------------------------------
// Host-side orchestration. H=1 => softmax over a single logit == 1, so
// attn_out == x@w_kv + b_kv exactly; w_attn/b_attn/mask are dead inputs.
// ---------------------------------------------------------------------------
extern "C" void kernel_launch(void* const* d_in, const int* in_sizes, int n_in,
                              void* d_out, int out_size, void* d_ws, size_t ws_size,
                              hipStream_t stream) {
    const float* x     = (const float*)d_in[0];
    const float* w_kv  = (const float*)d_in[4];
    const float* b_kv  = (const float*)d_in[5];
    const float* ln1_g = (const float*)d_in[6];
    const float* ln1_b = (const float*)d_in[7];
    const float* w_fc  = (const float*)d_in[8];
    const float* b_fc  = (const float*)d_in[9];
    const float* w_mp  = (const float*)d_in[10];
    const float* b_mp  = (const float*)d_in[11];
    const float* ln2_g = (const float*)d_in[12];
    const float* ln2_b = (const float*)d_in[13];
    float* out = (float*)d_out;

    unsigned char* ws = (unsigned char*)d_ws;
    size_t off = 0;
    auto carve = [&](size_t bytes) {
        void* p = ws + off;
        off += (bytes + 255) & ~(size_t)255;
        return p;
    };
    bf16*  xb   = (bf16*) carve((size_t)M_ROWS * E_DIM * 2);
    bf16*  wkvT = (bf16*) carve((size_t)E_DIM * E_DIM * 2);
    bf16*  wfcT = (bf16*) carve((size_t)FF_DIM * E_DIM * 2);
    bf16*  wmpT = (bf16*) carve((size_t)E_DIM * FF_DIM * 2);
    float* t1   = (float*)carve((size_t)M_ROWS * E_DIM * 4);   // also reused as t2
    bf16*  hb   = (bf16*) carve((size_t)M_ROWS * E_DIM * 2);
    bf16*  hm   = (bf16*) carve((size_t)M_ROWS * FF_DIM * 2);

    // 1) precision conversions (one-time, bandwidth-trivial vs GEMM time)
    {
        int n4 = M_ROWS * E_DIM / 4;
        cvt_f32_bf16<<<n4 / 256, 256, 0, stream>>>(x, xb, n4);
    }
    transpose_cvt<<<dim3(E_DIM / 32, E_DIM / 32), 256, 0, stream>>>(w_kv, wkvT, E_DIM, E_DIM);
    transpose_cvt<<<dim3(FF_DIM / 32, E_DIM / 32), 256, 0, stream>>>(w_fc, wfcT, E_DIM, FF_DIM);
    transpose_cvt<<<dim3(E_DIM / 32, FF_DIM / 32), 256, 0, stream>>>(w_mp, wmpT, FF_DIM, E_DIM);

    // 2) t1 = x + (x @ w_kv + b_kv)   [attention collapses to the kv projection]
    gemm_bf16<0><<<dim3(E_DIM / BN, M_ROWS / BM), 256, 0, stream>>>(
        xb, wkvT, b_kv, x, t1, nullptr, M_ROWS, E_DIM, E_DIM);

    // 3) hb = LN1(t1)  (bf16 for next GEMM)
    ln_kernel<true><<<M_ROWS, 256, 0, stream>>>(t1, ln1_g, ln1_b, nullptr, hb);

    // 4) hm = gelu(hb @ w_fc + b_fc)  (bf16)
    gemm_bf16<1><<<dim3(FF_DIM / BN, M_ROWS / BM), 256, 0, stream>>>(
        hb, wfcT, b_fc, nullptr, nullptr, hm, M_ROWS, FF_DIM, E_DIM);

    // 5) t2 = x + (hm @ w_mproj + b_mproj)  (reuse t1 buffer)
    gemm_bf16<0><<<dim3(E_DIM / BN, M_ROWS / BM), 256, 0, stream>>>(
        hm, wmpT, b_mp, x, t1, nullptr, M_ROWS, E_DIM, FF_DIM);

    // 6) out = LN2(t2)
    ln_kernel<false><<<M_ROWS, 256, 0, stream>>>(t1, ln2_g, ln2_b, out, nullptr);
}
